// MultiHeadAttention_64518998720697
// MI455X (gfx1250) — matrix operand order repaired and checked by reading, not compiled, not run on hardware
//
#include <hip/hip_runtime.h>
#include <hip/hip_bf16.h>

typedef __attribute__((ext_vector_type(16))) _Float16 v16h;
typedef __attribute__((ext_vector_type(8)))  _Float16 v8h;
typedef __attribute__((ext_vector_type(8)))  float    v8f;
typedef __attribute__((ext_vector_type(4)))  float    v4f;

#define D_MODEL 1024
#define N_HEADS 16
#define HEAD_DIM 64
#define SEQ 2048
#define BATCH 4
#define M_TOTAL (BATCH * SEQ) /* 8192 */

__device__ __forceinline__ v8f wmma_f16(v16h a, v16h b, v8f c) {
  return __builtin_amdgcn_wmma_f32_16x16x32_f16(false, a, false, b, (short)0, c,
                                                false, false);
}

// ---------------- weight transpose: W f32 [K,N] -> Wt f16 [N,K] -------------
__global__ void transpose_cast_kernel(const float* __restrict__ W,
                                      _Float16* __restrict__ Wt) {
  __shared__ float tile[32][33];
  const int bx = blockIdx.x * 32; // n base
  const int by = blockIdx.y * 32; // k base
  const int tx = threadIdx.x;     // 0..31
  const int ty = threadIdx.y;     // 0..7
#pragma unroll
  for (int i = 0; i < 32; i += 8)
    tile[ty + i][tx] = W[(size_t)(by + ty + i) * D_MODEL + bx + tx];
  __syncthreads();
#pragma unroll
  for (int i = 0; i < 32; i += 8)
    Wt[(size_t)(bx + ty + i) * D_MODEL + by + tx] = (_Float16)tile[tx][ty + i];
}

// ---------------- A-fragment helpers ---------------------------------------
struct ARaw { v4f r0, r1, r2, r3; };

__device__ __forceinline__ ARaw load_a_f32(const float* p) {
  ARaw x;
  x.r0 = *(const v4f*)(p);
  x.r1 = *(const v4f*)(p + 4);
  x.r2 = *(const v4f*)(p + 8);
  x.r3 = *(const v4f*)(p + 12);
  return x;
}

__device__ __forceinline__ v16h cvt_a(ARaw x) {
  v16h a;
#pragma unroll
  for (int i = 0; i < 4; ++i) {
    a[i]      = (_Float16)x.r0[i];
    a[4 + i]  = (_Float16)x.r1[i];
    a[8 + i]  = (_Float16)x.r2[i];
    a[12 + i] = (_Float16)x.r3[i];
  }
  return a;
}

struct AH { v8h lo, hi; };

__device__ __forceinline__ AH load_a_f16(const _Float16* p) {
  AH x;
  x.lo = *(const v8h*)(p);
  x.hi = *(const v8h*)(p + 8);
  return x;
}

__device__ __forceinline__ v16h cvt_ah(AH x) {
  v16h a;
#pragma unroll
  for (int i = 0; i < 8; ++i) { a[i] = x.lo[i]; a[8 + i] = x.hi[i]; }
  return a;
}

// ---------------- QKV projection: out[f16] = A[f32] @ Wt^T + bias ----------
// Wave tile 32x64, ping-pong double-buffered fragments (no rotation copies).
__global__ void __launch_bounds__(128, 1)
proj_gemm_kernel(const float* __restrict__ A,
                 const _Float16* __restrict__ Wt,
                 const float* __restrict__ bias,
                 _Float16* __restrict__ out,
                 int store_transposed_v) {
  const int lane = threadIdx.x & 31;
  const int wave = threadIdx.x >> 5;
  const int kh = lane >> 4;
  const int cl = lane & 15;
  const int m0 = blockIdx.x * 128 + wave * 32;
  const int n0 = blockIdx.y * 64;
  const float* arow0 = A + (size_t)(m0 + cl) * D_MODEL + kh * 16;
  const float* arow1 = arow0 + (size_t)16 * D_MODEL;
  const _Float16* wb = Wt + (size_t)(n0 + cl) * D_MODEL + kh * 16;

  v8f acc[2][4];
#pragma unroll
  for (int g = 0; g < 2; ++g)
#pragma unroll
    for (int j = 0; j < 4; ++j) acc[g][j] = (v8f){};

  ARaw a0x, a1x, a0y, a1y;
  v16h bx0, bx1, bx2, bx3, by0, by1, by2, by3;

  auto loadX = [&](int k0) {
    a0x = load_a_f32(arow0 + k0);
    a1x = load_a_f32(arow1 + k0);
    bx0 = *(const v16h*)(wb + k0);
    bx1 = *(const v16h*)(wb + k0 + 16 * D_MODEL);
    bx2 = *(const v16h*)(wb + k0 + 32 * D_MODEL);
    bx3 = *(const v16h*)(wb + k0 + 48 * D_MODEL);
  };
  auto loadY = [&](int k0) {
    a0y = load_a_f32(arow0 + k0);
    a1y = load_a_f32(arow1 + k0);
    by0 = *(const v16h*)(wb + k0);
    by1 = *(const v16h*)(wb + k0 + 16 * D_MODEL);
    by2 = *(const v16h*)(wb + k0 + 32 * D_MODEL);
    by3 = *(const v16h*)(wb + k0 + 48 * D_MODEL);
  };
  auto mma = [&](const ARaw& A0, const ARaw& A1, v16h b0, v16h b1, v16h b2,
                 v16h b3) {
    v16h va0 = cvt_a(A0);
    v16h va1 = cvt_a(A1);
    acc[0][0] = wmma_f16(va0, b0, acc[0][0]);
    acc[1][0] = wmma_f16(va1, b0, acc[1][0]);
    acc[0][1] = wmma_f16(va0, b1, acc[0][1]);
    acc[1][1] = wmma_f16(va1, b1, acc[1][1]);
    acc[0][2] = wmma_f16(va0, b2, acc[0][2]);
    acc[1][2] = wmma_f16(va1, b2, acc[1][2]);
    acc[0][3] = wmma_f16(va0, b3, acc[0][3]);
    acc[1][3] = wmma_f16(va1, b3, acc[1][3]);
  };

  loadX(0);
  for (int k0 = 0; k0 < D_MODEL; k0 += 64) {
    loadY(k0 + 32);                       // always in range (<= 992)
    mma(a0x, a1x, bx0, bx1, bx2, bx3);
    if (k0 + 64 < D_MODEL) loadX(k0 + 64);
    mma(a0y, a1y, by0, by1, by2, by3);
  }

#pragma unroll
  for (int g = 0; g < 2; ++g) {
#pragma unroll
    for (int j = 0; j < 4; ++j) {
      const int ncol = n0 + j * 16 + cl;
      const float bvv = bias[ncol];
#pragma unroll
      for (int r = 0; r < 8; ++r) {
        const int mrow = m0 + g * 16 + r + kh * 8;
        const float val = acc[g][j][r] + bvv;
        if (store_transposed_v) {
          const int bb = mrow >> 11, s = mrow & (SEQ - 1);
          const int hh = ncol >> 6, d = ncol & 63;
          out[((size_t)((bb * N_HEADS + hh) * HEAD_DIM + d) << 11) + s] =
              (_Float16)val;
        } else {
          out[(size_t)mrow * D_MODEL + ncol] = (_Float16)val;
        }
      }
    }
  }
}

// ---------------- flash attention ------------------------------------------
// One wave = 16 query rows of one (b,h). Online softmax; ping-pong K frags.
__global__ void __launch_bounds__(128, 1)
flash_attn_kernel(const _Float16* __restrict__ Q,
                  const _Float16* __restrict__ K,
                  const _Float16* __restrict__ Vt,
                  _Float16* __restrict__ Ctx) {
  __shared__ _Float16 pbuf[4][16 * 32];
  const int lane = threadIdx.x & 31;
  const int wave = threadIdx.x >> 5;
  const int kh = lane >> 4;
  const int cl = lane & 15;
  const int bh = blockIdx.y;
  const int b = bh >> 4;
  const int h = bh & 15;
  const int q0 = blockIdx.x * 64 + wave * 16;

  const _Float16* qbase =
      Q + (size_t)(b * SEQ + q0 + cl) * D_MODEL + h * HEAD_DIM;
  v16h aq0 = cvt_ah(load_a_f16(qbase + kh * 16));
  v16h aq1 = cvt_ah(load_a_f16(qbase + 32 + kh * 16));

  float mrow[8], lsum[8];
#pragma unroll
  for (int r = 0; r < 8; ++r) { mrow[r] = -3.0e38f; lsum[r] = 0.0f; }
  v8f c0 = {}, c1 = {}, c2 = {}, c3 = {};
  _Float16* pb = &pbuf[wave][0];
  const _Float16* kcol =
      K + (size_t)(b * SEQ + cl) * D_MODEL + h * HEAD_DIM + kh * 16;
  const _Float16* vbh = Vt + ((size_t)((b * N_HEADS + h) * HEAD_DIM) << 11);
  const float scale = 0.125f; // 1/sqrt(64)

  auto loadK = [&](int kc, v16h* f) {
    const _Float16* p = kcol + (size_t)kc * D_MODEL;
    f[0] = *(const v16h*)(p);
    f[1] = *(const v16h*)(p + 32);
    f[2] = *(const v16h*)(p + 16 * D_MODEL);
    f[3] = *(const v16h*)(p + 16 * D_MODEL + 32);
  };

  // one 32-key chunk: scores -> online softmax -> LDS transpose -> P@V
  auto chunk = [&](int kc, const v16h* kf) {
    const _Float16* vp = vbh + ((size_t)cl << 11) + kc + kh * 16;
    v16h vf0 = *(const v16h*)(vp);
    v16h vf1 = *(const v16h*)(vp + (16 << 11));
    v16h vf2 = *(const v16h*)(vp + (32 << 11));
    v16h vf3 = *(const v16h*)(vp + (48 << 11));

    v8f s0 = {}, s1 = {};
    s0 = wmma_f16(aq0, kf[0], s0);
    s0 = wmma_f16(aq1, kf[1], s0);
    s1 = wmma_f16(aq0, kf[2], s1);
    s1 = wmma_f16(aq1, kf[3], s1);

    float t[8], e0[8], e1[8];
#pragma unroll
    for (int r = 0; r < 8; ++r) {
      e0[r] = s0[r] * scale;
      e1[r] = s1[r] * scale;
      t[r] = fmaxf(e0[r], e1[r]);
    }
#pragma unroll
    for (int off = 8; off >= 1; off >>= 1) {
#pragma unroll
      for (int r = 0; r < 8; ++r)
        t[r] = fmaxf(t[r], __shfl_xor(t[r], off, 16));
    }
#pragma unroll
    for (int r = 0; r < 8; ++r) {
      const float mnew = fmaxf(mrow[r], t[r]);
      const float alpha = __expf(mrow[r] - mnew);
      mrow[r] = mnew;
      const float p0 = __expf(e0[r] - mnew);
      const float p1 = __expf(e1[r] - mnew);
      float ps = p0 + p1;
#pragma unroll
      for (int off = 8; off >= 1; off >>= 1) ps += __shfl_xor(ps, off, 16);
      lsum[r] = lsum[r] * alpha + ps;
      c0[r] *= alpha; c1[r] *= alpha; c2[r] *= alpha; c3[r] *= alpha;
      const int prow = r + kh * 8; // C-layout row -> LDS row
      pb[prow * 32 + cl] = (_Float16)p0;
      pb[prow * 32 + 16 + cl] = (_Float16)p1;
    }
    __builtin_amdgcn_wave_barrier(); // intra-wave LDS ordering; DS in-order

    v16h pa = cvt_ah(load_a_f16(pb + cl * 32 + kh * 16));

    c0 = wmma_f16(pa, vf0, c0);
    c1 = wmma_f16(pa, vf1, c1);
    c2 = wmma_f16(pa, vf2, c2);
    c3 = wmma_f16(pa, vf3, c3);
  };

  v16h kfx[4], kfy[4];
  loadK(0, kfx);
  for (int kc = 0; kc < SEQ; kc += 64) {
    loadK(kc + 32, kfy);                  // always in range (<= 2016)
    __builtin_prefetch(kcol + (size_t)(kc + 64) * D_MODEL, 0, 1);
    chunk(kc, kfx);
    if (kc + 64 < SEQ) loadK(kc + 64, kfx);
    chunk(kc + 32, kfy);
  }

  v8f cc[4] = {c0, c1, c2, c3};
#pragma unroll
  for (int r = 0; r < 8; ++r) {
    const float inv = 1.0f / lsum[r];
    const size_t orow =
        (size_t)(b * SEQ + q0 + r + kh * 8) * D_MODEL + h * HEAD_DIM;
#pragma unroll
    for (int j = 0; j < 4; ++j)
      Ctx[orow + j * 16 + cl] = (_Float16)(cc[j][r] * inv);
  }
}

// ---------------- output projection: out[f32] = Ctx[f16] @ Wo^T + bo -------
__global__ void __launch_bounds__(128, 1)
out_proj_kernel(const _Float16* __restrict__ A,
                const _Float16* __restrict__ Wt,
                const float* __restrict__ bias,
                float* __restrict__ out) {
  const int lane = threadIdx.x & 31;
  const int wave = threadIdx.x >> 5;
  const int kh = lane >> 4;
  const int cl = lane & 15;
  const int m0 = blockIdx.x * 128 + wave * 32;
  const int n0 = blockIdx.y * 64;
  const _Float16* arow0 = A + (size_t)(m0 + cl) * D_MODEL + kh * 16;
  const _Float16* arow1 = arow0 + (size_t)16 * D_MODEL;
  const _Float16* wb = Wt + (size_t)(n0 + cl) * D_MODEL + kh * 16;

  v8f acc[2][4];
#pragma unroll
  for (int g = 0; g < 2; ++g)
#pragma unroll
    for (int j = 0; j < 4; ++j) acc[g][j] = (v8f){};

  AH a0x, a1x, a0y, a1y;
  v16h bx0, bx1, bx2, bx3, by0, by1, by2, by3;

  auto loadX = [&](int k0) {
    a0x = load_a_f16(arow0 + k0);
    a1x = load_a_f16(arow1 + k0);
    bx0 = *(const v16h*)(wb + k0);
    bx1 = *(const v16h*)(wb + k0 + 16 * D_MODEL);
    bx2 = *(const v16h*)(wb + k0 + 32 * D_MODEL);
    bx3 = *(const v16h*)(wb + k0 + 48 * D_MODEL);
  };
  auto loadY = [&](int k0) {
    a0y = load_a_f16(arow0 + k0);
    a1y = load_a_f16(arow1 + k0);
    by0 = *(const v16h*)(wb + k0);
    by1 = *(const v16h*)(wb + k0 + 16 * D_MODEL);
    by2 = *(const v16h*)(wb + k0 + 32 * D_MODEL);
    by3 = *(const v16h*)(wb + k0 + 48 * D_MODEL);
  };
  auto mma = [&](const AH& A0, const AH& A1, v16h b0, v16h b1, v16h b2,
                 v16h b3) {
    v16h va0 = cvt_ah(A0);
    v16h va1 = cvt_ah(A1);
    acc[0][0] = wmma_f16(va0, b0, acc[0][0]);
    acc[1][0] = wmma_f16(va1, b0, acc[1][0]);
    acc[0][1] = wmma_f16(va0, b1, acc[0][1]);
    acc[1][1] = wmma_f16(va1, b1, acc[1][1]);
    acc[0][2] = wmma_f16(va0, b2, acc[0][2]);
    acc[1][2] = wmma_f16(va1, b2, acc[1][2]);
    acc[0][3] = wmma_f16(va0, b3, acc[0][3]);
    acc[1][3] = wmma_f16(va1, b3, acc[1][3]);
  };

  loadX(0);
  for (int k0 = 0; k0 < D_MODEL; k0 += 64) {
    loadY(k0 + 32);
    mma(a0x, a1x, bx0, bx1, bx2, bx3);
    if (k0 + 64 < D_MODEL) loadX(k0 + 64);
    mma(a0y, a1y, by0, by1, by2, by3);
  }

#pragma unroll
  for (int g = 0; g < 2; ++g) {
#pragma unroll
    for (int j = 0; j < 4; ++j) {
      const int ncol = n0 + j * 16 + cl;
      const float bvv = bias[ncol];
#pragma unroll
      for (int r = 0; r < 8; ++r) {
        const int mrow = m0 + g * 16 + r + kh * 8;
        out[(size_t)mrow * D_MODEL + ncol] = acc[g][j][r] + bvv;
      }
    }
  }
}

extern "C" void kernel_launch(void* const* d_in, const int* in_sizes, int n_in,
                              void* d_out, int out_size, void* d_ws,
                              size_t ws_size, hipStream_t stream) {
  const float* q = (const float*)d_in[0];
  const float* k = (const float*)d_in[1];
  const float* v = (const float*)d_in[2];
  const float* Wq = (const float*)d_in[3];
  const float* bq = (const float*)d_in[4];
  const float* Wk = (const float*)d_in[5];
  const float* bk = (const float*)d_in[6];
  const float* Wv = (const float*)d_in[7];
  const float* bv = (const float*)d_in[8];
  const float* Wo = (const float*)d_in[9];
  const float* bo = (const float*)d_in[10];
  float* out = (float*)d_out;

  char* ws = (char*)d_ws;
  size_t off = 0;
  const size_t wbytes = (size_t)D_MODEL * D_MODEL * sizeof(_Float16); // 2 MB
  const size_t mbytes = (size_t)M_TOTAL * D_MODEL * sizeof(_Float16); // 16 MB
  _Float16* WqT = (_Float16*)(ws + off); off += wbytes;
  _Float16* WkT = (_Float16*)(ws + off); off += wbytes;
  _Float16* WvT = (_Float16*)(ws + off); off += wbytes;
  _Float16* WoT = (_Float16*)(ws + off); off += wbytes;
  _Float16* Qp  = (_Float16*)(ws + off); off += mbytes;
  _Float16* Kp  = (_Float16*)(ws + off); off += mbytes;
  _Float16* Vt  = (_Float16*)(ws + off); off += mbytes;
  _Float16* Ctx = (_Float16*)(ws + off); off += mbytes;
  (void)ws_size; (void)in_sizes; (void)n_in; (void)out_size;

  dim3 tcb(32, 8), tcg(D_MODEL / 32, D_MODEL / 32);
  transpose_cast_kernel<<<tcg, tcb, 0, stream>>>(Wq, WqT);
  transpose_cast_kernel<<<tcg, tcb, 0, stream>>>(Wk, WkT);
  transpose_cast_kernel<<<tcg, tcb, 0, stream>>>(Wv, WvT);
  transpose_cast_kernel<<<tcg, tcb, 0, stream>>>(Wo, WoT);

  dim3 gb(128); // 4 waves, each a 32x64 tile
  dim3 gg(M_TOTAL / 128, D_MODEL / 64);
  proj_gemm_kernel<<<gg, gb, 0, stream>>>(q, WqT, bq, Qp, 0);
  proj_gemm_kernel<<<gg, gb, 0, stream>>>(k, WkT, bk, Kp, 0);
  proj_gemm_kernel<<<gg, gb, 0, stream>>>(v, WvT, bv, Vt, 1);

  dim3 ag(SEQ / 64, BATCH * N_HEADS);
  flash_attn_kernel<<<ag, gb, 0, stream>>>(Qp, Kp, Vt, Ctx);

  out_proj_kernel<<<gg, gb, 0, stream>>>(Ctx, WoT, bo, out);
}
